// FINNv2_7971459301397
// MI455X (gfx1250) — 8-bit matrix operand order repaired and checked by reading, compiled, not run on hardware
//
#include <hip/hip_runtime.h>
#include <cstdint>
#include <cstddef>

typedef __attribute__((ext_vector_type(8))) int v8i;
typedef __attribute__((ext_vector_type(4))) int v4i;

static inline int ceil_div(int a, int b) { return (a + b - 1) / b; }

__device__ __forceinline__ void async_ld_b128(uint32_t lds_off, const void* g) {
  asm volatile("global_load_async_to_lds_b128 %0, %1, off"
               :: "v"(lds_off), "v"(g) : "memory");
}
__device__ __forceinline__ void wait_async0() {
  asm volatile("s_wait_asynccnt 0x0" ::: "memory");
}


__global__ void k_quant_input(const float* __restrict__ x, int8_t* __restrict__ out,
                              int B, int C, int H, int W) {
  int idx = blockIdx.x * blockDim.x + threadIdx.x;
  int n = B * C * H * W;
  if (idx >= n) return;
  int ci = idx % C; int t = idx / C;
  int xx = t % W; t /= W;
  int yy = t % H; int b = t / H;
  float v = x[((b * C + ci) * H + yy) * W + xx];
  float q = rintf((2.0f * v - 1.0f) * 128.0f);
  q = fminf(fmaxf(q, -128.0f), 127.0f);
  out[idx] = (int8_t)(int)q;
}

__global__ void k_pack_w(const float* __restrict__ w, int8_t* __restrict__ wp,
                         int Cout, int Cin, int KH, int KW, int K, int Kpad) {
  int idx = blockIdx.x * blockDim.x + threadIdx.x;
  int n = Cout * Kpad;
  if (idx >= n) return;
  int kp = idx % Kpad; int co = idx / Kpad;
  int8_t v = 0;
  if (kp < K) {
    int ci = kp % Cin; int pos = kp / Cin;
    int ky = pos / KW; int kx = pos % KW;
    float f = w[(((size_t)co * Cin + ci) * KH + ky) * KW + kx];
    float q = fminf(fmaxf(rintf(f), -1.0f), 1.0f);
    v = (int8_t)(int)q;
  }
  wp[idx] = v;
}

__global__ void k_prep_bn(const float* __restrict__ g, const float* __restrict__ b,
                          const float* __restrict__ m, const float* __restrict__ v,
                          float a_scale, float* __restrict__ scale,
                          float* __restrict__ shift, int C) {
  int i = blockIdx.x * blockDim.x + threadIdx.x;
  if (i >= C) return;
  float s = g[i] * rsqrtf(v[i] + 1e-4f);
  scale[i] = a_scale * s;
  shift[i] = b[i] - m[i] * s;
}

__global__ void k_prep_final(const float* __restrict__ tnw, const float* __restrict__ tnb,
                             float* __restrict__ scale, float* __restrict__ shift, int C) {
  int i = blockIdx.x * blockDim.x + threadIdx.x;
  if (i >= C) return;
  scale[i] = 0.5f * tnw[0];
  shift[i] = tnb[0];
}

__global__ void k_pool2x2(const int8_t* __restrict__ in, int8_t* __restrict__ out,
                          int B, int H, int W, int C) {
  int idx = blockIdx.x * blockDim.x + threadIdx.x;
  int Ho = H >> 1, Wo = W >> 1;
  int n = B * Ho * Wo * C;
  if (idx >= n) return;
  int c = idx % C; int t = idx / C;
  int ox = t % Wo; t /= Wo;
  int oy = t % Ho; int b = t / Ho;
  const int8_t* p = in + (((size_t)b * H + 2 * oy) * W + 2 * ox) * C + c;
  int v0 = p[0], v1 = p[C], v2 = p[(size_t)W * C], v3 = p[(size_t)W * C + C];
  int mx = v0 > v1 ? v0 : v1;
  mx = v2 > mx ? v2 : mx;
  mx = v3 > mx ? v3 : mx;
  out[idx] = (int8_t)mx;
}

__device__ __forceinline__ v8i fragA64(const uint32_t* row, int half) {
  v8i r; int o = half * 4;
  r[0] = row[o + 0];  r[1] = row[o + 1];
  r[2] = row[o + 2];  r[3] = row[o + 3];
  r[4] = row[o + 8];  r[5] = row[o + 9];
  r[6] = row[o + 10]; r[7] = row[o + 11];
  return r;
}
__device__ __forceinline__ v8i fragB64(const uint32_t* row, int half) {
  v8i r; int o = half * 4;
  r[0] = row[o + 0]; r[1] = row[o + 1]; r[2] = row[o + 2]; r[3] = row[o + 3];
  r[4] = row[o + 8]; r[5] = row[o + 9]; r[6] = row[o + 10]; r[7] = row[o + 11];
  return r;
}

template <bool FINAL>
__global__ __launch_bounds__(256) void k_gemm_async(
    const int8_t* __restrict__ A, const int8_t* __restrict__ Wp,
    const float* __restrict__ scale, const float* __restrict__ shift,
    int8_t* __restrict__ outq, float* __restrict__ outf,
    int M, int Cout, int Hin, int Win, int Cin,
    int Hout, int Wout, int KW, int Kpad) {
  __shared__ uint8_t At[2][128 * 64];
  __shared__ uint8_t Bt[2][64 * 64];

  const int tid = threadIdx.x;
  const int lane = tid & 31;
  const int wave = tid >> 5;
  const int m0 = blockIdx.x * 128;
  const int n0 = blockIdx.y * 64;

  const int ar = tid >> 1;
  const int apart = tid & 1;
  const int am = m0 + ar;
  const bool am_ok = am < M;
  int ab = 0, aoy = 0, aox = 0;
  if (am_ok) {
    int hw = Hout * Wout;
    ab = am / hw; int r2 = am - ab * hw;
    aoy = r2 / Wout; aox = r2 - aoy * Wout;
  }
  const int brow = tid >> 2;
  const int bpart = tid & 3;
  const int bn = n0 + brow;
  const bool bn_ok = bn < Cout;

  auto issue = [&](int k0, int buf) {
    if (am_ok) {
      int pos = k0 / Cin;
      int ci0 = k0 - pos * Cin;
      int ky = pos / KW, kx = pos - ky * KW;
      const int8_t* gp = A + ((((size_t)ab * Hin + (aoy + ky)) * Win + (aox + kx)) * Cin)
                           + ci0 + apart * 32;
      uint32_t l = (uint32_t)(uintptr_t)&At[buf][ar * 64 + apart * 32];
      async_ld_b128(l, gp);
      async_ld_b128(l + 16, gp + 16);
    }
    if (bn_ok) {
      uint32_t l = (uint32_t)(uintptr_t)&Bt[buf][brow * 64 + bpart * 16];
      async_ld_b128(l, Wp + (size_t)bn * Kpad + k0 + bpart * 16);
    }
  };

  const v8i zero = {0, 0, 0, 0, 0, 0, 0, 0};
  v8i acc[4];
#pragma unroll
  for (int s = 0; s < 4; ++s) acc[s] = zero;

  const int half = lane >> 4;
  const int l16 = lane & 15;

  issue(0, 0);
  int buf = 0;
  for (int k0 = 0; k0 < Kpad; k0 += 64) {
    wait_async0();
    __syncthreads();
    if (k0 + 64 < Kpad) issue(k0 + 64, buf ^ 1);

    v8i a = fragA64((const uint32_t*)&At[buf][((wave << 4) + l16) * 64], half);
#pragma unroll
    for (int s = 0; s < 4; ++s) {
      v8i b = fragB64((const uint32_t*)&Bt[buf][(s * 16 + l16) * 64], half);
      acc[s] = __builtin_amdgcn_wmma_i32_16x16x64_iu8(true, a, true, b, acc[s], false, false);
    }
    buf ^= 1;
  }

  const int mBase = m0 + (wave << 4) + ((lane >> 4) << 3);
  const int nl = lane & 15;
#pragma unroll
  for (int s = 0; s < 4; ++s) {
    int n = n0 + s * 16 + nl;
    if (n >= Cout) continue;
    float sc = scale[n], sh = shift[n];
#pragma unroll
    for (int r = 0; r < 8; ++r) {
      int m = mBase + r;
      if (m >= M) continue;
      float y = (float)acc[s][r] * sc + sh;
      if (FINAL) {
        outf[(size_t)m * Cout + n] = y;
      } else {
        int q = (int)rintf(y * 2.0f);
        q = q < -2 ? -2 : (q > 1 ? 1 : q);
        outq[(size_t)m * Cout + n] = (int8_t)q;
      }
    }
  }
}

__global__ __launch_bounds__(256) void k_gemm_gather(
    const int8_t* __restrict__ A, const int8_t* __restrict__ Wp,
    const float* __restrict__ scale, const float* __restrict__ shift,
    int8_t* __restrict__ outq,
    int M, int Cout, int Hin, int Win, int Cin,
    int Hout, int Wout, int KW, int K, int Kpad) {
  __shared__ uint8_t At[128 * 64];
  __shared__ uint8_t Bt[64 * 64];

  const int tid = threadIdx.x;
  const int lane = tid & 31;
  const int wave = tid >> 5;
  const int m0 = blockIdx.x * 128;
  const int n0 = blockIdx.y * 64;

  const int ar = tid >> 1;
  const int apart = tid & 1;
  const int am = m0 + ar;
  const bool am_ok = am < M;
  int ab = 0, aoy = 0, aox = 0;
  if (am_ok) {
    int hw = Hout * Wout;
    ab = am / hw; int r2 = am - ab * hw;
    aoy = r2 / Wout; aox = r2 - aoy * Wout;
  }
  const int brow = tid >> 2;
  const int bpart = tid & 3;
  const int bn = n0 + brow;
  const bool bn_ok = bn < Cout;

  const v8i zero = {0, 0, 0, 0, 0, 0, 0, 0};
  v8i acc[4];
#pragma unroll
  for (int s = 0; s < 4; ++s) acc[s] = zero;

  for (int k0 = 0; k0 < Kpad; k0 += 64) {
#pragma unroll 4
    for (int j = 0; j < 32; ++j) {
      int kk = apart * 32 + j;
      int k = k0 + kk;
      int8_t v = 0;
      if (am_ok && k < K) {
        int pos = k / Cin; int ci = k - pos * Cin;
        int ky = pos / KW, kx = pos - ky * KW;
        v = A[(((size_t)ab * Hin + (aoy + ky)) * Win + (aox + kx)) * Cin + ci];
      }
      At[ar * 64 + kk] = v;
    }
    {
      v4i d = {0, 0, 0, 0};
      if (bn_ok) d = *(const v4i*)(Wp + (size_t)bn * Kpad + k0 + bpart * 16);
      *(v4i*)&Bt[brow * 64 + bpart * 16] = d;
    }
    __syncthreads();

    const int half = lane >> 4;
    const int l16 = lane & 15;
    v8i a = fragA64((const uint32_t*)&At[((wave << 4) + l16) * 64], half);
#pragma unroll
    for (int s = 0; s < 4; ++s) {
      v8i b = fragB64((const uint32_t*)&Bt[(s * 16 + l16) * 64], half);
      acc[s] = __builtin_amdgcn_wmma_i32_16x16x64_iu8(true, a, true, b, acc[s], false, false);
    }
    __syncthreads();
  }

  const int mBase = m0 + (wave << 4) + ((lane >> 4) << 3);
  const int nl = lane & 15;
#pragma unroll
  for (int s = 0; s < 4; ++s) {
    int n = n0 + s * 16 + nl;
    if (n >= Cout) continue;
    float sc = scale[n], sh = shift[n];
#pragma unroll
    for (int r = 0; r < 8; ++r) {
      int m = mBase + r;
      if (m >= M) continue;
      float y = (float)acc[s][r] * sc + sh;
      int q = (int)rintf(y * 2.0f);
      q = q < -2 ? -2 : (q > 1 ? 1 : q);
      outq[(size_t)m * Cout + n] = (int8_t)q;
    }
  }
}


extern "C" void kernel_launch(void* const* d_in, const int* in_sizes, int n_in,
                              void* d_out, int out_size, void* d_ws, size_t ws_size,
                              hipStream_t stream) {
  (void)in_sizes; (void)n_in; (void)out_size; (void)ws_size;
  const float* x = (const float*)d_in[0];
  const float *w[6], *g[6], *bb[6], *mm[6], *vv[6];
  for (int i = 0; i < 6; ++i) {
    w[i]  = (const float*)d_in[1 + 5 * i];
    g[i]  = (const float*)d_in[2 + 5 * i];
    bb[i] = (const float*)d_in[3 + 5 * i];
    mm[i] = (const float*)d_in[4 + 5 * i];
    vv[i] = (const float*)d_in[5 + 5 * i];
  }
  const float* fw[3] = {(const float*)d_in[31], (const float*)d_in[32], (const float*)d_in[33]};
  const float* fg[2] = {(const float*)d_in[34], (const float*)d_in[38]};
  const float* fb[2] = {(const float*)d_in[35], (const float*)d_in[39]};
  const float* fm[2] = {(const float*)d_in[36], (const float*)d_in[40]};
  const float* fv[2] = {(const float*)d_in[37], (const float*)d_in[41]};
  const float* tnw = (const float*)d_in[42];
  const float* tnb = (const float*)d_in[43];
  float* out = (float*)d_out;

  char* base = (char*)d_ws;
  size_t off = 0;
  auto alloc = [&](size_t bytes) -> void* {
    void* p = base + off; off += (bytes + 255) & ~(size_t)255; return p;
  };
  const int B = 512;
  int8_t* xq = (int8_t*)alloc((size_t)B * 32 * 32 * 3);
  int8_t* a0 = (int8_t*)alloc((size_t)B * 30 * 30 * 64);
  int8_t* a1 = (int8_t*)alloc((size_t)B * 28 * 28 * 64);
  int8_t* p1 = (int8_t*)alloc((size_t)B * 14 * 14 * 64);
  int8_t* a2 = (int8_t*)alloc((size_t)B * 12 * 12 * 128);
  int8_t* a3 = (int8_t*)alloc((size_t)B * 10 * 10 * 128);
  int8_t* p3 = (int8_t*)alloc((size_t)B * 5 * 5 * 128);
  int8_t* a4 = (int8_t*)alloc((size_t)B * 3 * 3 * 256);
  int8_t* a5 = (int8_t*)alloc((size_t)B * 256);
  int8_t* f0 = (int8_t*)alloc((size_t)B * 512);
  int8_t* f1 = (int8_t*)alloc((size_t)B * 512);
  int8_t* w0p = (int8_t*)alloc(64 * 64);
  int8_t* w1p = (int8_t*)alloc((size_t)64 * 576);
  int8_t* w2p = (int8_t*)alloc((size_t)128 * 576);
  int8_t* w3p = (int8_t*)alloc((size_t)128 * 1152);
  int8_t* w4p = (int8_t*)alloc((size_t)256 * 1152);
  int8_t* w5p = (int8_t*)alloc((size_t)256 * 2304);
  int8_t* f0p = (int8_t*)alloc((size_t)512 * 256);
  int8_t* f1p = (int8_t*)alloc((size_t)512 * 512);
  int8_t* f2p = (int8_t*)alloc((size_t)10 * 512);
  float* sc[9]; float* sh[9];
  const int cch[9] = {64, 64, 128, 128, 256, 256, 512, 512, 10};
  for (int i = 0; i < 9; ++i) {
    sc[i] = (float*)alloc(cch[i] * sizeof(float));
    sh[i] = (float*)alloc(cch[i] * sizeof(float));
  }

  const float IN_SCALE = 0.0078125f;
  const float ACT_SCALE = 0.5f;

  { int n = B * 3 * 32 * 32;
    k_quant_input<<<ceil_div(n, 256), 256, 0, stream>>>(x, xq, B, 3, 32, 32); }

  const int cinl[6]  = {3, 64, 64, 128, 128, 256};
  const int coutl[6] = {64, 64, 128, 128, 256, 256};
  const int kl[6]    = {27, 576, 576, 1152, 1152, 2304};
  const int kpl[6]   = {64, 576, 576, 1152, 1152, 2304};
  int8_t* wp[6] = {w0p, w1p, w2p, w3p, w4p, w5p};
  for (int i = 0; i < 6; ++i) {
    int n = coutl[i] * kpl[i];
    k_pack_w<<<ceil_div(n, 256), 256, 0, stream>>>(w[i], wp[i], coutl[i], cinl[i], 3, 3, kl[i], kpl[i]);
    k_prep_bn<<<ceil_div(coutl[i], 256), 256, 0, stream>>>(
        g[i], bb[i], mm[i], vv[i], i == 0 ? IN_SCALE : ACT_SCALE, sc[i], sh[i], coutl[i]);
  }
  k_pack_w<<<ceil_div(512 * 256, 256), 256, 0, stream>>>(fw[0], f0p, 512, 256, 1, 1, 256, 256);
  k_pack_w<<<ceil_div(512 * 512, 256), 256, 0, stream>>>(fw[1], f1p, 512, 512, 1, 1, 512, 512);
  k_pack_w<<<ceil_div(10 * 512, 256), 256, 0, stream>>>(fw[2], f2p, 10, 512, 1, 1, 512, 512);
  for (int i = 0; i < 2; ++i)
    k_prep_bn<<<2, 256, 0, stream>>>(fg[i], fb[i], fm[i], fv[i], ACT_SCALE, sc[6 + i], sh[6 + i], 512);
  k_prep_final<<<1, 256, 0, stream>>>(tnw, tnb, sc[8], sh[8], 10);

  { int M = B * 30 * 30;
    k_gemm_gather<<<dim3(ceil_div(M, 128), 1), 256, 0, stream>>>(
        xq, w0p, sc[0], sh[0], a0, M, 64, 32, 32, 3, 30, 30, 3, 27, 64); }
  { int M = B * 28 * 28;
    k_gemm_async<false><<<dim3(ceil_div(M, 128), 1), 256, 0, stream>>>(
        a0, w1p, sc[1], sh[1], a1, nullptr, M, 64, 30, 30, 64, 28, 28, 3, 576); }
  { int n = B * 14 * 14 * 64;
    k_pool2x2<<<ceil_div(n, 256), 256, 0, stream>>>(a1, p1, B, 28, 28, 64); }
  { int M = B * 12 * 12;
    k_gemm_async<false><<<dim3(ceil_div(M, 128), 2), 256, 0, stream>>>(
        p1, w2p, sc[2], sh[2], a2, nullptr, M, 128, 14, 14, 64, 12, 12, 3, 576); }
  { int M = B * 10 * 10;
    k_gemm_async<false><<<dim3(ceil_div(M, 128), 2), 256, 0, stream>>>(
        a2, w3p, sc[3], sh[3], a3, nullptr, M, 128, 12, 12, 128, 10, 10, 3, 1152); }
  { int n = B * 5 * 5 * 128;
    k_pool2x2<<<ceil_div(n, 256), 256, 0, stream>>>(a3, p3, B, 10, 10, 128); }
  { int M = B * 3 * 3;
    k_gemm_async<false><<<dim3(ceil_div(M, 128), 4), 256, 0, stream>>>(
        p3, w4p, sc[4], sh[4], a4, nullptr, M, 256, 5, 5, 128, 3, 3, 3, 1152); }
  { int M = B;
    k_gemm_async<false><<<dim3(ceil_div(M, 128), 4), 256, 0, stream>>>(
        a4, w5p, sc[5], sh[5], a5, nullptr, M, 256, 3, 3, 256, 1, 1, 3, 2304); }
  k_gemm_async<false><<<dim3(4, 8), 256, 0, stream>>>(
      a5, f0p, sc[6], sh[6], f0, nullptr, B, 512, 1, 1, 256, 1, 1, 1, 256);
  k_gemm_async<false><<<dim3(4, 8), 256, 0, stream>>>(
      f0, f1p, sc[7], sh[7], f1, nullptr, B, 512, 1, 1, 512, 1, 1, 1, 512);
  k_gemm_async<true><<<dim3(4, 1), 256, 0, stream>>>(
      f1, f2p, sc[8], sh[8], nullptr, out, B, 10, 1, 1, 512, 1, 1, 1, 512);
}
